// StructuralEncoder_1975684956487
// MI455X (gfx1250) — matrix operand order repaired and checked by reading, not compiled, not run on hardware
//
#include <hip/hip_runtime.h>
#include <hip/hip_bf16.h>

// ---------------------------------------------------------------------------
// StructuralEncoder for MI455X (gfx1250), wave32 + WMMA bf16 + async-to-LDS.
//   phase 0: convert entity/relation tables to bf16 (L2-resident, 51.2MB)
//            and W1/W2/W3 to bf16 transposed [N][K] (WMMA B layout)
//   phase 1: persistent blocks; per 128-row tile:
//              wait async x -> layer1 -> issue async gather of NEXT tile
//              -> layer2 -> layer3 reduced into per-lane registers
//   phase 2: deterministic reduce of per-block partials + head MLP -> d_out
// Fallback (small ws): f32 gather with in-kernel conversion, same layers.
// ---------------------------------------------------------------------------

typedef __attribute__((ext_vector_type(16))) __bf16 v16bf;
typedef __attribute__((ext_vector_type(8)))  __bf16 v8bf;
typedef __attribute__((ext_vector_type(8)))  float  v8f;
typedef __attribute__((ext_vector_type(4)))  float  v4f;

#define NBLOCKS 512
#define MTILE   128

union ABfrag { v16bf v; v8bf h[2]; };

__device__ __forceinline__ __bf16 tobf(float f) { return (__bf16)f; }

struct Smem {
    __bf16 x[MTILE * 384];    // 98304 B : gathered input tile (layer1 A)
    __bf16 h1[MTILE * 256];   // 65536 B : layer1 out / layer2 A
    __bf16 h2[MTILE * 256];   // 65536 B : layer2 out / layer3 A
    float  bias[640];         //  2560 B : b1(256) | b2(256) | b3(128)
    float  red[256];          //  1024 B : final per-lane reduction
};

// Generic WMMA layer: out[128][N] = relu(in[128][K] @ Wt^T + bias)
// Wt is stored transposed: [N][K] bf16 in global (L2-resident).
// bias points into LDS (preloaded) -> ds_load_b32 in the nt prologue.
template<int K, int N, bool RELU>
__device__ __forceinline__ void layer_fwd(const __bf16* __restrict__ in,
                                          const __bf16* __restrict__ Wt,
                                          const float*  __restrict__ bias,
                                          __bf16* __restrict__ out,
                                          int tid)
{
    const int lane  = tid & 31;
    const int wave  = tid >> 5;
    const int arow  = lane & 15;                 // A: M row
    const int akofs = (lane & 16) ? 8 : 0;       // A: K offset for hi lanes
    const int bkofs = (lane & 16) ? 8 : 0;       // B: K offset for hi lanes
    const int rbase = (lane & 16) ? 8 : 0;       // C/D: row offset for hi lanes

    for (int nt = wave; nt < N / 16; nt += 8) {
        const int ncol = nt * 16 + (lane & 15);
        const float bv = bias[ncol];
        v8f acc[8];
        #pragma unroll
        for (int m = 0; m < 8; ++m) {
            #pragma unroll
            for (int r = 0; r < 8; ++r) acc[m][r] = bv;
        }
        const __bf16* wrow = Wt + (size_t)ncol * K + bkofs;
        #pragma unroll 2        // cap pipelining window: no spills, 16-WMMA span
        for (int k0 = 0; k0 < K; k0 += 32) {
            ABfrag b;
            b.h[0] = *(const v8bf*)(wrow + k0);          // K = bkofs+0..7
            b.h[1] = *(const v8bf*)(wrow + k0 + 16);     // K = bkofs+16..23
            #pragma unroll
            for (int m = 0; m < 8; ++m) {
                const __bf16* ap = in + (m * 16 + arow) * K + k0 + akofs;
                ABfrag a;
                a.h[0] = *(const v8bf*)(ap);             // K = akofs+0..7
                a.h[1] = *(const v8bf*)(ap + 16);        // K = akofs+16..23
                acc[m] = __builtin_amdgcn_wmma_f32_16x16x32_bf16(
                    false, a.v, false, b.v, (short)0, acc[m], false, false);
            }
        }
        #pragma unroll
        for (int m = 0; m < 8; ++m) {
            #pragma unroll
            for (int r = 0; r < 8; ++r) {
                float v = acc[m][r];
                if (RELU) v = v > 0.0f ? v : 0.0f;
                out[(m * 16 + rbase + r) * N + ncol] = tobf(v);
            }
        }
    }
}

// Layer 3: enc tile = in[128][256] @ W3t^T + b3; returns this lane's masked
// row-sum for its fixed output column (ncol = wave*16 + (lane&15)).
__device__ __forceinline__ float layer3_partial(const __bf16* __restrict__ in,
                                                const __bf16* __restrict__ Wt,
                                                const float*  __restrict__ bias,
                                                int tilebase, int nrows, int tid)
{
    constexpr int K = 256;
    const int lane  = tid & 31;
    const int wave  = tid >> 5;
    const int arow  = lane & 15;
    const int akofs = (lane & 16) ? 8 : 0;
    const int bkofs = (lane & 16) ? 8 : 0;
    const int rbase = (lane & 16) ? 8 : 0;

    const int ncol = wave * 16 + (lane & 15);   // N=128: one 16-col tile/wave
    const float bv = bias[ncol];
    v8f acc[8];
    #pragma unroll
    for (int m = 0; m < 8; ++m) {
        #pragma unroll
        for (int r = 0; r < 8; ++r) acc[m][r] = bv;
    }
    const __bf16* wrow = Wt + (size_t)ncol * K + bkofs;
    #pragma unroll 2
    for (int k0 = 0; k0 < K; k0 += 32) {
        ABfrag b;
        b.h[0] = *(const v8bf*)(wrow + k0);
        b.h[1] = *(const v8bf*)(wrow + k0 + 16);
        #pragma unroll
        for (int m = 0; m < 8; ++m) {
            const __bf16* ap = in + (m * 16 + arow) * K + k0 + akofs;
            ABfrag a;
            a.h[0] = *(const v8bf*)(ap);
            a.h[1] = *(const v8bf*)(ap + 16);
            acc[m] = __builtin_amdgcn_wmma_f32_16x16x32_bf16(
                false, a.v, false, b.v, (short)0, acc[m], false, false);
        }
    }
    float s = 0.0f;
    #pragma unroll
    for (int m = 0; m < 8; ++m) {
        #pragma unroll
        for (int r = 0; r < 8; ++r) {
            const int grow = tilebase + m * 16 + rbase + r;
            s += (grow < nrows) ? acc[m][r] : 0.0f;   // tail-row mask
        }
    }
    return s;
}

// Async gather (bf16 tables): copy 256B/row-segment global -> LDS, no VGPR
// round trip, tracked by ASYNCcnt. IOFFSET applies to BOTH the LDS and the
// global address, so one base pair + 16 immediate offsets covers a segment.
__device__ __forceinline__ void issue_gather_async(Smem& sm,
                                                   const int* __restrict__ triples,
                                                   const __bf16* __restrict__ entB,
                                                   const __bf16* __restrict__ relB,
                                                   int tilebase, int nrows, int tid)
{
    for (int seg = tid; seg < MTILE * 3; seg += 256) {
        const int row  = seg / 3;
        const int part = seg - row * 3;
        const int grow = tilebase + row;
        if (grow >= nrows) continue;         // stale rows masked in layer3
        const int idx = triples[(size_t)grow * 3 + part];
        const __bf16* g = (part == 1 ? relB : entB) + (size_t)idx * 128;
        const unsigned ldsa =
            (unsigned)(uintptr_t)(void*)(sm.x + row * 384 + part * 128);
        #define GASYNC(OFF)                                               \
            asm volatile("global_load_async_to_lds_b128 %0, %1, off offset:" #OFF \
                         :: "v"(ldsa), "v"(g) : "memory")
        GASYNC(0);   GASYNC(16);  GASYNC(32);  GASYNC(48);
        GASYNC(64);  GASYNC(80);  GASYNC(96);  GASYNC(112);
        GASYNC(128); GASYNC(144); GASYNC(160); GASYNC(176);
        GASYNC(192); GASYNC(208); GASYNC(224); GASYNC(240);
        #undef GASYNC
    }
}

// Fallback gather: f32 tables, convert in-kernel, ds_store.
__device__ __forceinline__ void gather_f32(Smem& sm,
                                           const int* __restrict__ triples,
                                           const float* __restrict__ entity,
                                           const float* __restrict__ relation,
                                           int tilebase, int nrows, int tid)
{
    for (int seg = tid; seg < MTILE * 3; seg += 256) {
        const int row  = seg / 3;
        const int part = seg - row * 3;
        const int grow = tilebase + row;
        if (grow >= nrows) continue;
        const int idx = triples[(size_t)grow * 3 + part];
        const float* src = (part == 1 ? relation : entity) + (size_t)idx * 128;
        __bf16* dst = sm.x + row * 384 + part * 128;
        #pragma unroll
        for (int j = 0; j < 16; ++j) {
            v4f f0 = *(const v4f*)(src + j * 8);
            v4f f1 = *(const v4f*)(src + j * 8 + 4);
            v8bf o = { tobf(f0[0]), tobf(f0[1]), tobf(f0[2]), tobf(f0[3]),
                       tobf(f1[0]), tobf(f1[1]), tobf(f1[2]), tobf(f1[3]) };
            *(v8bf*)(dst + j * 8) = o;
        }
    }
}

template<bool ASYNC>
__global__ __launch_bounds__(256)
void encoder_main(const int*   __restrict__ triples,
                  const float* __restrict__ entityF,
                  const float* __restrict__ relationF,
                  const __bf16* __restrict__ entB,
                  const __bf16* __restrict__ relB,
                  const float* __restrict__ b1,
                  const float* __restrict__ b2,
                  const float* __restrict__ b3,
                  const __bf16* __restrict__ W1t,
                  const __bf16* __restrict__ W2t,
                  const __bf16* __restrict__ W3t,
                  float* __restrict__ partials,
                  int nrows)
{
    __shared__ Smem sm;
    const int tid    = threadIdx.x;
    const int stride = gridDim.x;
    const int ntiles = (nrows + MTILE - 1) / MTILE;
    float csum = 0.0f;   // this lane's running column sum (fixed ncol)

    // preload biases into LDS: removes per-nt global loads from hot loop
    for (int i = tid; i < 640; i += 256) {
        sm.bias[i] = (i < 256) ? b1[i]
                   : (i < 512) ? b2[i - 256]
                               : b3[i - 512];
    }

    if (ASYNC && (int)blockIdx.x < ntiles)
        issue_gather_async(sm, triples, entB, relB, blockIdx.x * MTILE,
                           nrows, tid);

    for (int tile = blockIdx.x; tile < ntiles; tile += stride) {
        const int tilebase = tile * MTILE;
        if (ASYNC) {
            asm volatile("s_wait_asynccnt 0x0" ::: "memory");
        } else {
            gather_f32(sm, triples, entityF, relationF, tilebase, nrows, tid);
        }
        __syncthreads();                                  // x(tile)+bias ready

        layer_fwd<384, 256, true>(sm.x, W1t, sm.bias, sm.h1, tid);
        __syncthreads();                                  // x now dead

        if (ASYNC && tile + stride < ntiles)              // prefetch next x
            issue_gather_async(sm, triples, entB, relB,
                               (tile + stride) * MTILE, nrows, tid);

        layer_fwd<256, 256, true>(sm.h1, W2t, sm.bias + 256, sm.h2, tid);
        __syncthreads();

        csum += layer3_partial(sm.h2, W3t, sm.bias + 512, tilebase, nrows, tid);
    }

    __syncthreads();
    sm.red[tid] = csum;
    __syncthreads();
    if (tid < 128) {
        const int w = tid >> 4, c = tid & 15;   // ncol=tid held by lanes c,c+16
        partials[(size_t)blockIdx.x * 128 + tid] =
            sm.red[w * 32 + c] + sm.red[w * 32 + c + 16];
    }
}

// ---- table conversion: f32 -> bf16, 8 elems/thread ----
__global__ __launch_bounds__(256)
void convert_table(const float* __restrict__ src, __bf16* __restrict__ dst, int n)
{
    const int i = (blockIdx.x * 256 + threadIdx.x) * 8;
    if (i + 8 <= n) {
        v4f f0 = *(const v4f*)(src + i);
        v4f f1 = *(const v4f*)(src + i + 4);
        v8bf o = { tobf(f0[0]), tobf(f0[1]), tobf(f0[2]), tobf(f0[3]),
                   tobf(f1[0]), tobf(f1[1]), tobf(f1[2]), tobf(f1[3]) };
        *(v8bf*)(dst + i) = o;
    }
}

// ---- weight conversion: f32 [K][N] -> bf16 transposed [N][K] ----
__global__ __launch_bounds__(256)
void convert_weights(const float* __restrict__ W1,
                     const float* __restrict__ W2,
                     const float* __restrict__ W3,
                     __bf16* __restrict__ W1t,
                     __bf16* __restrict__ W2t,
                     __bf16* __restrict__ W3t)
{
    const int i = blockIdx.x * 256 + threadIdx.x;
    if (i < 98304) {                       // W1: [384][256] -> W1t[256][384]
        const int n = i / 384, k = i - n * 384;
        W1t[i] = tobf(W1[k * 256 + n]);
    }
    const int j = i - 98304;
    if (j >= 0 && j < 65536) {             // W2: [256][256] -> W2t[256][256]
        const int n = j / 256, k = j - n * 256;
        W2t[j] = tobf(W2[k * 256 + n]);
    }
    const int l = i - 163840;
    if (l >= 0 && l < 32768) {             // W3: [256][128] -> W3t[128][256]
        const int n = l / 256, k = l - n * 256;
        W3t[l] = tobf(W3[k * 128 + n]);
    }
}

// ---- deterministic reduce + head MLP ----
__global__ __launch_bounds__(256)
void head_kernel(const float* __restrict__ partials, int nblocks,
                 const float* __restrict__ Wf1, const float* __restrict__ bf1,
                 const float* __restrict__ Wf2, const float* __restrict__ bf2,
                 float* __restrict__ out, float inv_n)
{
    __shared__ float agg[128];
    __shared__ float t[256];
    const int tid = threadIdx.x;
    if (tid < 128) {
        float s = 0.0f;
        for (int b = 0; b < nblocks; ++b) s += partials[(size_t)b * 128 + tid];
        agg[tid] = s * inv_n;
    }
    __syncthreads();
    {   // t[j] = relu(bf1[j] + sum_k agg[k] * Wf1[k][j]),  Wf1: [128][256]
        float s = bf1[tid];
        for (int k = 0; k < 128; ++k) s += agg[k] * Wf1[k * 256 + tid];
        t[tid] = s > 0.0f ? s : 0.0f;
    }
    __syncthreads();
    if (tid < 128) {  // out[i] = bf2[i] + sum_j t[j] * Wf2[j][i], Wf2: [256][128]
        float s = bf2[tid];
        for (int j = 0; j < 256; ++j) s += t[j] * Wf2[j * 128 + tid];
        out[tid] = s;
    }
}

extern "C" void kernel_launch(void* const* d_in, const int* in_sizes, int n_in,
                              void* d_out, int out_size, void* d_ws, size_t ws_size,
                              hipStream_t stream) {
    const int*   triples  = (const int*)  d_in[0];
    const float* entity   = (const float*)d_in[1];
    const float* relation = (const float*)d_in[2];
    const float* W1  = (const float*)d_in[3];
    const float* b1  = (const float*)d_in[4];
    const float* W2  = (const float*)d_in[5];
    const float* b2  = (const float*)d_in[6];
    const float* W3  = (const float*)d_in[7];
    const float* b3  = (const float*)d_in[8];
    const float* Wf1 = (const float*)d_in[9];
    const float* bf1 = (const float*)d_in[10];
    const float* Wf2 = (const float*)d_in[11];
    const float* bf2 = (const float*)d_in[12];
    float* out = (float*)d_out;

    const int nrows = in_sizes[0] / 3;
    const int ntab1 = in_sizes[1];     // entity table elems (VOCAB*128)
    const int ntab2 = in_sizes[2];     // relation table elems

    // workspace layout (bytes):
    //   [0, 262144)        : partials  NBLOCKS x 128 f32
    //   [262144, +196608)  : W1t bf16 [256][384]
    //   [458752, +131072)  : W2t bf16 [256][256]
    //   [589824, +65536)   : W3t bf16 [128][256]
    //   [655360, +2*ntab1) : entity table bf16   (async path only)
    //   [.. ,   +2*ntab2)  : relation table bf16 (async path only)
    char* ws = (char*)d_ws;
    float*  partials = (float*) (ws);
    __bf16* W1t      = (__bf16*)(ws + 262144);
    __bf16* W2t      = (__bf16*)(ws + 458752);
    __bf16* W3t      = (__bf16*)(ws + 589824);
    __bf16* entB     = (__bf16*)(ws + 655360);
    __bf16* relB     = entB + ntab1;
    const size_t need_async =
        655360 + 2ull * (size_t)ntab1 + 2ull * (size_t)ntab2;
    const bool use_async = ws_size >= need_async;
    (void)n_in; (void)out_size;

    convert_weights<<<768, 256, 0, stream>>>(W1, W2, W3, W1t, W2t, W3t);
    if (use_async) {
        convert_table<<<(ntab1 / 8 + 255) / 256, 256, 0, stream>>>(entity, entB, ntab1);
        convert_table<<<(ntab2 / 8 + 255) / 256, 256, 0, stream>>>(relation, relB, ntab2);
        encoder_main<true><<<NBLOCKS, 256, 0, stream>>>(
            triples, entity, relation, entB, relB,
            b1, b2, b3, W1t, W2t, W3t, partials, nrows);
    } else {
        encoder_main<false><<<NBLOCKS, 256, 0, stream>>>(
            triples, entity, relation, nullptr, nullptr,
            b1, b2, b3, W1t, W2t, W3t, partials, nrows);
    }
    head_kernel<<<1, 256, 0, stream>>>(partials, NBLOCKS, Wf1, bf1, Wf2, bf2,
                                       out, 1.0f / (float)nrows);
}
